// EmbedMatcher_90220083020358
// MI455X (gfx1250) — matrix operand order repaired and checked by reading, not compiled, not run on hardware
//
#include <hip/hip_runtime.h>
#include <hip/hip_bf16.h>
#include <math.h>

#define NBR      200
#define EMBED    128
#define DMODEL   256
#define DINNER   512
#define LSTMH    512
#define NGATES   2048
#define BATCH    2048
#define SUP_PAD  16
#define LN_EPSF  1e-3f

typedef __bf16 bf16_t;
typedef __attribute__((ext_vector_type(16))) __bf16 v16bf;
typedef __attribute__((ext_vector_type(8)))  float v8f;

// ---------------------------------------------------------------------------
// Gather-sum: out[row, slot*128+d] = sum_k emb[conn[row,k,slot], d]
// ---------------------------------------------------------------------------
__global__ void gather_sum_kernel(const int* __restrict__ conn,
                                  const float* __restrict__ emb,
                                  float* __restrict__ out) {
    const int row  = blockIdx.x;
    const int t    = threadIdx.x;
    const int slot = t >> 7;     // 0: rel, 1: ent
    const int d    = t & 127;
    const int* cr = conn + (size_t)row * NBR * 2;
    float acc = 0.f;
    for (int k = 0; k < NBR; ++k) {
        int idx = cr[k * 2 + slot];
        acc += emb[(size_t)idx * EMBED + d];
    }
    out[(size_t)row * (2 * EMBED) + slot * EMBED + d] = acc;
}

// ---------------------------------------------------------------------------
// GCN row matvec: dst[row, colOff+t] = tanh((S[row]·W[t] + NBR*b[t]) / deg[row])
// ---------------------------------------------------------------------------
__global__ void gcn_rowmv_kernel(const float* __restrict__ S,
                                 const float* __restrict__ deg,
                                 const float* __restrict__ W,      // 128 x 256
                                 const float* __restrict__ bias,   // 128
                                 float* __restrict__ dst,
                                 int colOff, int ldd, int rowsValid) {
    __shared__ float s[2 * EMBED];
    const int row = blockIdx.x;
    const int t   = threadIdx.x;      // 0..127
    s[t]       = S[(size_t)row * (2 * EMBED) + t];
    s[t + 128] = S[(size_t)row * (2 * EMBED) + t + 128];
    __syncthreads();
    const float* wr = W + (size_t)t * (2 * EMBED);
    float a = 0.f;
    #pragma unroll 4
    for (int c = 0; c < 2 * EMBED; ++c) a += s[c] * wr[c];
    float dv = (row < rowsValid) ? deg[row] : 1.0f;
    a = (a + (float)NBR * bias[t]) / dv;
    float v = (row < rowsValid) ? tanhf(a) : 0.f;
    dst[(size_t)row * ldd + colOff + t] = v;
}

// ---------------------------------------------------------------------------
// Pack weight W (Nrows x Kcols row-major f32) -> bf16 WMMA B-fragment order:
// tiles (K/32 x N/16), within tile: lane = (n%16)+16*(wk>>4), elem = wk&15.
// ---------------------------------------------------------------------------
__global__ void pack_b_kernel(const float* __restrict__ W,
                              bf16_t* __restrict__ Bw,
                              int Nrows, int Kcols) {
    int i = blockIdx.x * blockDim.x + threadIdx.x;
    if (i >= Nrows * Kcols) return;
    int n = i / Kcols, k = i % Kcols;
    int kt = k >> 5, wk = k & 31;
    int half = wk >> 4, e = wk & 15;
    int nt = n >> 4, ln = n & 15;
    size_t dst = ((size_t)kt * (Nrows >> 4) + nt) * 512
               + (size_t)((ln + (half << 4)) * 16 + e);
    Bw[dst] = (bf16_t)W[i];
}

// ---------------------------------------------------------------------------
// Pack activation X (M x K row-major f32) -> bf16 WMMA A-fragment order:
// tiles (M/16 x K/32) with K innermost, within tile:
//   lane = (m%16) + 16*(wk>>4), elem = wk&15, wk = k%32.
// ---------------------------------------------------------------------------
__device__ __forceinline__ size_t a_frag_index(int m, int k, int Ktiles) {
    int wk = k & 31;
    int half = wk >> 4;
    int e = wk & 15;
    int lane = (m & 15) + (half << 4);
    return ((size_t)(m >> 4) * Ktiles + (k >> 5)) * 512 + (size_t)(lane * 16 + e);
}

__global__ void pack_a_kernel(const float* __restrict__ X,
                              bf16_t* __restrict__ Aw,
                              int M, int K) {
    int i = blockIdx.x * blockDim.x + threadIdx.x;
    if (i >= M * K) return;
    int m = i / K, k = i % K;
    Aw[a_frag_index(m, k, K >> 5)] = (bf16_t)X[i];
}

__global__ void add_vec_kernel(const float* __restrict__ a,
                               const float* __restrict__ b,
                               float* __restrict__ o, int n) {
    int i = blockIdx.x * blockDim.x + threadIdx.x;
    if (i < n) o[i] = a[i] + b[i];
}

__global__ void fill_zero_bf16_kernel(bf16_t* __restrict__ p, int n) {
    int i = blockIdx.x * blockDim.x + threadIdx.x;
    if (i < n) p[i] = (bf16_t)0.f;
}

__global__ void fill_zero_f32_kernel(float* __restrict__ p, int n) {
    int i = blockIdx.x * blockDim.x + threadIdx.x;
    if (i < n) p[i] = 0.f;
}

// ---------------------------------------------------------------------------
// One K-sweep: acc[0..3] += A(16xKp) x B(Kpx64).  Both operands pre-packed
// in fragment order; per chunk: 2 b128 loads (A), 8 b128 loads (B), 4 WMMAs.
// No LDS, no barriers, no conversions in the loop.
// ---------------------------------------------------------------------------
__device__ __forceinline__ void gemm_phase(const bf16_t* __restrict__ Aw, int Kp,
                                           const bf16_t* __restrict__ Bw,
                                           int Ntiles, int ntBase, int mt,
                                           int lane, v8f acc[4]) {
    const int Ktiles = Kp >> 5;
    const bf16_t* ap = Aw + ((size_t)mt * Ktiles) * 512 + lane * 16;
    for (int kt = 0; kt < Ktiles; ++kt) {
        v16bf a = *(const v16bf*)(ap + (size_t)kt * 512);
        const bf16_t* bbase =
            Bw + ((size_t)kt * Ntiles + ntBase) * 512 + lane * 16;
        #pragma unroll
        for (int t = 0; t < 4; ++t) {
            v16bf b = *(const v16bf*)(bbase + (size_t)t * 512);
            acc[t] = __builtin_amdgcn_wmma_f32_16x16x32_bf16(
                false, a, false, b, (short)0, acc[t], false, false);
        }
    }
}

// ---------------------------------------------------------------------------
// Tiled GEMM: C = epilogue( A@B [+ A2@B2] + bias ).  Block = 128 thr (4
// waves); block tile 16(M) x 256(N); wave owns 4 16x16 tiles.
// MODE: 0 bias, 1 bias+relu, 2 bias+residual.
// ---------------------------------------------------------------------------
template <int MODE, bool DUAL>
__global__ void wmma_gemm_kernel(const bf16_t* __restrict__ Aw, int K,
                                 const bf16_t* __restrict__ Bw,
                                 const bf16_t* __restrict__ A2w, int K2,
                                 const bf16_t* __restrict__ B2w,
                                 const float* __restrict__ bias,
                                 const float* __restrict__ residual, int ldr,
                                 float* __restrict__ C, int ldc, int N) {
    const int wave   = threadIdx.x >> 5;
    const int lane   = threadIdx.x & 31;
    const int mt     = blockIdx.y;
    const int ntBase = blockIdx.x * 16 + wave * 4;
    const int Ntiles = N >> 4;

    v8f acc[4] = {};
    gemm_phase(Aw, K, Bw, Ntiles, ntBase, mt, lane, acc);
    if (DUAL)
        gemm_phase(A2w, K2, B2w, Ntiles, ntBase, mt, lane, acc);

    const int nloc = lane & 15;
    const int mOff = (lane >> 4) << 3;
    #pragma unroll
    for (int t = 0; t < 4; ++t) {
        const int n = (ntBase + t) * 16 + nloc;
        const float bv = bias[n];
        #pragma unroll
        for (int r = 0; r < 8; ++r) {
            const int m = mt * 16 + mOff + r;
            float v = acc[t][r] + bv;
            if (MODE == 1) v = fmaxf(v, 0.f);
            if (MODE == 2) v += residual[(size_t)m * ldr + n];
            C[(size_t)m * ldc + n] = v;
        }
    }
}

// ---------------------------------------------------------------------------
// LayerNorm (ddof=1) over rows of 256.
// ---------------------------------------------------------------------------
__global__ void layernorm_kernel(const float* __restrict__ X,
                                 const float* __restrict__ ga,
                                 const float* __restrict__ gb,
                                 float* __restrict__ Y) {
    __shared__ float rs[DMODEL], rq[DMODEL];
    const int row = blockIdx.x, t = threadIdx.x;
    float x = X[(size_t)row * DMODEL + t];
    rs[t] = x; rq[t] = x * x;
    __syncthreads();
    for (int s = DMODEL / 2; s > 0; s >>= 1) {
        if (t < s) { rs[t] += rs[t + s]; rq[t] += rq[t + s]; }
        __syncthreads();
    }
    float mu    = rs[0] / (float)DMODEL;
    float var   = (rq[0] - (float)DMODEL * mu * mu) / (float)(DMODEL - 1);
    float sigma = sqrtf(fmaxf(var, 0.f));
    Y[(size_t)row * DMODEL + t] = (x - mu) / (sigma + LN_EPSF) * ga[t] + gb[t];
}

__global__ void support_mean_kernel(const float* __restrict__ sg,
                                    float* __restrict__ sm) {
    int t = threadIdx.x;   // 256
    float a = 0.f;
    for (int r = 0; r < 5; ++r) a += sg[(size_t)r * DMODEL + t];
    sm[t] = a * 0.2f;
}

// ---------------------------------------------------------------------------
// LSTM pointwise; softmax over singleton axis == 1 => r = support_mean.
// Writes next h_r DIRECTLY in packed bf16 A-fragment order (no repack pass).
// ---------------------------------------------------------------------------
__global__ void lstm_pointwise_kernel(const float* __restrict__ gates,
                                      float* __restrict__ c,
                                      const float* __restrict__ qg,
                                      const float* __restrict__ smean,
                                      float* __restrict__ h,
                                      bf16_t* __restrict__ hr_pk) {
    int i = blockIdx.x * blockDim.x + threadIdx.x;
    if (i >= BATCH * LSTMH) return;
    int row = i / LSTMH, col = i % LSTMH;
    const float* g = gates + (size_t)row * NGATES;
    float ig = 1.f / (1.f + expf(-g[col]));
    float fg = 1.f / (1.f + expf(-g[LSTMH + col]));
    float gg = tanhf(g[2 * LSTMH + col]);
    float og = 1.f / (1.f + expf(-g[3 * LSTMH + col]));
    float cn = fg * c[i] + ig * gg;
    c[i] = cn;
    float hn = og * tanhf(cn);
    float hrv;
    if (col < DMODEL) {
        float hv = qg[(size_t)row * DMODEL + col] + hn;
        h[(size_t)row * DMODEL + col] = hv;
        hrv = hv;
    } else {
        hrv = smean[col - DMODEL];
    }
    hr_pk[a_frag_index(row, col, LSTMH >> 5)] = (bf16_t)hrv;
}

__global__ void score_dot_kernel(const float* __restrict__ h,
                                 const float* __restrict__ smean,
                                 float* __restrict__ out) {
    __shared__ float rs[DMODEL];
    const int row = blockIdx.x, t = threadIdx.x;
    rs[t] = h[(size_t)row * DMODEL + t] * smean[t];
    __syncthreads();
    for (int s = DMODEL / 2; s > 0; s >>= 1) {
        if (t < s) rs[t] += rs[t + s];
        __syncthreads();
    }
    if (t == 0) out[row] = rs[0];
}

// ---------------------------------------------------------------------------
extern "C" void kernel_launch(void* const* d_in, const int* in_sizes, int n_in,
                              void* d_out, int out_size, void* d_ws, size_t ws_size,
                              hipStream_t stream) {
    const int*   qlc  = (const int*)  d_in[0];
    const float* qld  = (const float*)d_in[1];
    const int*   qrc  = (const int*)  d_in[2];
    const float* qrd  = (const float*)d_in[3];
    const int*   slc  = (const int*)  d_in[4];
    const float* sld  = (const float*)d_in[5];
    const int*   src  = (const int*)  d_in[6];
    const float* srd  = (const float*)d_in[7];
    const float* emb  = (const float*)d_in[8];
    const float* gcnw = (const float*)d_in[9];
    const float* gcnb = (const float*)d_in[10];
    const float* p1w  = (const float*)d_in[11];
    const float* p1b  = (const float*)d_in[12];
    const float* p2w  = (const float*)d_in[13];
    const float* p2b  = (const float*)d_in[14];
    const float* lna  = (const float*)d_in[15];
    const float* lnb  = (const float*)d_in[16];
    const float* wih  = (const float*)d_in[17];
    const float* whh  = (const float*)d_in[18];
    const float* bih  = (const float*)d_in[19];
    const float* bhh  = (const float*)d_in[20];
    float* out = (float*)d_out;

    char* ws = (char*)d_ws;
    size_t off = 0;
    auto carve = [&](size_t bytes) -> char* {
        char* p = ws + off;
        off = (off + bytes + 255) & ~(size_t)255;
        return p;
    };
    float*  sumL_q = (float*) carve((size_t)BATCH * 2 * EMBED * 4);
    float*  sumR_q = (float*) carve((size_t)BATCH * 2 * EMBED * 4);
    float*  sumL_s = (float*) carve((size_t)SUP_PAD * 2 * EMBED * 4);
    float*  sumR_s = (float*) carve((size_t)SUP_PAD * 2 * EMBED * 4);
    float*  qn     = (float*) carve((size_t)BATCH * DMODEL * 4);
    float*  sn     = (float*) carve((size_t)SUP_PAD * DMODEL * 4);
    float*  h1_q   = (float*) carve((size_t)BATCH * DINNER * 4);
    float*  h1_s   = (float*) carve((size_t)SUP_PAD * DINNER * 4);
    float*  ffn_q  = (float*) carve((size_t)BATCH * DMODEL * 4);
    float*  ffn_s  = (float*) carve((size_t)SUP_PAD * DMODEL * 4);
    float*  qg     = (float*) carve((size_t)BATCH * DMODEL * 4);
    float*  sg     = (float*) carve((size_t)SUP_PAD * DMODEL * 4);
    float*  smean  = (float*) carve((size_t)DMODEL * 4);
    float*  cst    = (float*) carve((size_t)BATCH * LSTMH * 4);
    float*  hbuf   = (float*) carve((size_t)BATCH * DMODEL * 4);
    float*  gates  = (float*) carve((size_t)BATCH * NGATES * 4);
    // packed bf16 operands
    bf16_t* wih_w  = (bf16_t*)carve((size_t)DMODEL * NGATES * 2);
    bf16_t* whh_w  = (bf16_t*)carve((size_t)LSTMH  * NGATES * 2);
    bf16_t* p1sw   = (bf16_t*)carve((size_t)DMODEL * DINNER * 2);
    bf16_t* p2sw   = (bf16_t*)carve((size_t)DINNER * DMODEL * 2);
    bf16_t* qn_pk  = (bf16_t*)carve((size_t)BATCH * DMODEL * 2);
    bf16_t* sn_pk  = (bf16_t*)carve((size_t)SUP_PAD * DMODEL * 2);
    bf16_t* h1q_pk = (bf16_t*)carve((size_t)BATCH * DINNER * 2);
    bf16_t* h1s_pk = (bf16_t*)carve((size_t)SUP_PAD * DINNER * 2);
    bf16_t* qg_pk  = (bf16_t*)carve((size_t)BATCH * DMODEL * 2);
    bf16_t* hr_pk  = (bf16_t*)carve((size_t)BATCH * LSTMH * 2);
    float*  biasc  = (float*) carve((size_t)NGATES * 4);

    // --- weight prep: pack to bf16 WMMA fragment layouts ---
    pack_b_kernel<<<(NGATES * DMODEL + 255) / 256, 256, 0, stream>>>(wih, wih_w, NGATES, DMODEL);
    pack_b_kernel<<<(NGATES * LSTMH  + 255) / 256, 256, 0, stream>>>(whh, whh_w, NGATES, LSTMH);
    pack_b_kernel<<<(DINNER * DMODEL + 255) / 256, 256, 0, stream>>>(p1w, p1sw, DINNER, DMODEL);
    pack_b_kernel<<<(DMODEL * DINNER + 255) / 256, 256, 0, stream>>>(p2w, p2sw, DMODEL, DINNER);
    add_vec_kernel<<<(NGATES + 255) / 256, 256, 0, stream>>>(bih, bhh, biasc, NGATES);

    // --- neighbor encoder: gather-sum then GCN matvec+tanh ---
    gather_sum_kernel<<<BATCH, 256, 0, stream>>>(qlc, emb, sumL_q);
    gather_sum_kernel<<<BATCH, 256, 0, stream>>>(qrc, emb, sumR_q);
    gather_sum_kernel<<<5,     256, 0, stream>>>(slc, emb, sumL_s);
    gather_sum_kernel<<<5,     256, 0, stream>>>(src, emb, sumR_s);

    gcn_rowmv_kernel<<<BATCH,   128, 0, stream>>>(sumL_q, qld, gcnw, gcnb, qn, 0,     DMODEL, BATCH);
    gcn_rowmv_kernel<<<BATCH,   128, 0, stream>>>(sumR_q, qrd, gcnw, gcnb, qn, EMBED, DMODEL, BATCH);
    gcn_rowmv_kernel<<<SUP_PAD, 128, 0, stream>>>(sumL_s, sld, gcnw, gcnb, sn, 0,     DMODEL, 5);
    gcn_rowmv_kernel<<<SUP_PAD, 128, 0, stream>>>(sumR_s, srd, gcnw, gcnb, sn, EMBED, DMODEL, 5);

    pack_a_kernel<<<(BATCH * DMODEL + 255) / 256, 256, 0, stream>>>(qn, qn_pk, BATCH, DMODEL);
    pack_a_kernel<<<(SUP_PAD * DMODEL + 255) / 256, 256, 0, stream>>>(sn, sn_pk, SUP_PAD, DMODEL);

    // --- support encoder (FFN + residual + LN) ---
    {   // query: M=2048
        wmma_gemm_kernel<1, false><<<dim3(DINNER / 256, BATCH / 16), 128, 0, stream>>>(
            qn_pk, DMODEL, p1sw, nullptr, 0, nullptr,
            p1b, nullptr, 0, h1_q, DINNER, DINNER);
        pack_a_kernel<<<(BATCH * DINNER + 255) / 256, 256, 0, stream>>>(h1_q, h1q_pk, BATCH, DINNER);
        wmma_gemm_kernel<2, false><<<dim3(DMODEL / 256, BATCH / 16), 128, 0, stream>>>(
            h1q_pk, DINNER, p2sw, nullptr, 0, nullptr,
            p2b, qn, DMODEL, ffn_q, DMODEL, DMODEL);
        layernorm_kernel<<<BATCH, DMODEL, 0, stream>>>(ffn_q, lna, lnb, qg);
        pack_a_kernel<<<(BATCH * DMODEL + 255) / 256, 256, 0, stream>>>(qg, qg_pk, BATCH, DMODEL);
    }
    {   // support: M=16 (rows 5..15 inert padding)
        wmma_gemm_kernel<1, false><<<dim3(DINNER / 256, SUP_PAD / 16), 128, 0, stream>>>(
            sn_pk, DMODEL, p1sw, nullptr, 0, nullptr,
            p1b, nullptr, 0, h1_s, DINNER, DINNER);
        pack_a_kernel<<<(SUP_PAD * DINNER + 255) / 256, 256, 0, stream>>>(h1_s, h1s_pk, SUP_PAD, DINNER);
        wmma_gemm_kernel<2, false><<<dim3(DMODEL / 256, SUP_PAD / 16), 128, 0, stream>>>(
            h1s_pk, DINNER, p2sw, nullptr, 0, nullptr,
            p2b, sn, DMODEL, ffn_s, DMODEL, DMODEL);
        layernorm_kernel<<<SUP_PAD, DMODEL, 0, stream>>>(ffn_s, lna, lnb, sg);
    }
    support_mean_kernel<<<1, DMODEL, 0, stream>>>(sg, smean);

    // --- LSTM matching processor: zero initial h_r (packed) and c state ---
    fill_zero_bf16_kernel<<<(BATCH * LSTMH + 255) / 256, 256, 0, stream>>>(hr_pk, BATCH * LSTMH);
    fill_zero_f32_kernel<<<(BATCH * LSTMH + 255) / 256, 256, 0, stream>>>(cst, BATCH * LSTMH);

    for (int step = 0; step < 4; ++step) {
        // gates = qg @ wih^T + h_r @ whh^T + (b_ih + b_hh)   [fused dual GEMM]
        wmma_gemm_kernel<0, true><<<dim3(NGATES / 256, BATCH / 16), 128, 0, stream>>>(
            qg_pk, DMODEL, wih_w, hr_pk, LSTMH, whh_w,
            biasc, nullptr, 0, gates, NGATES, NGATES);
        lstm_pointwise_kernel<<<(BATCH * LSTMH + 255) / 256, 256, 0, stream>>>(
            gates, cst, qg, smean, hbuf, hr_pk);
    }

    // --- matching scores ---
    score_dot_kernel<<<BATCH, DMODEL, 0, stream>>>(hbuf, smean, out);
}
